// ReferenceGQA_59657095741550
// MI455X (gfx1250) — compile-verified, matrix operand order checked by reading, not run on hardware
//
#include <hip/hip_runtime.h>

typedef _Float16 half_t;
typedef __attribute__((ext_vector_type(16))) _Float16 v16h;
typedef __attribute__((ext_vector_type(8)))  _Float16 v8h;
typedef __attribute__((ext_vector_type(8)))  float    v8f;

namespace {

constexpr int Bb   = 2;
constexpr int Nn   = 2048;
constexpr int Dd   = 2048;
constexpr int Hh   = 32;
constexpr int KVH  = 8;
constexpr int HD   = 64;
constexpr int MTOK = Bb * Nn;                // 4096 tokens
constexpr int QKVN = Hh * HD + 2 * KVH * HD; // 3072 fused output cols

// 1/sqrt(64) * log2(e): folded into wq so attention can use exp2 directly.
constexpr float QSCALE = 0.125f * 1.4426950408889634f;

__device__ __forceinline__ v16h frag16(const half_t* p) {
  // A/B fragment per ISA: lane holds 8 contiguous f16 at base and 8 at base+16.
  v8h lo = *(const v8h*)(p);
  v8h hi = *(const v8h*)(p + 16);
  return __builtin_shufflevector(lo, hi, 0,1,2,3,4,5,6,7,8,9,10,11,12,13,14,15);
}

__device__ __forceinline__ v8f wmma_f16(v16h a, v16h b, v8f c) {
  return __builtin_amdgcn_wmma_f32_16x16x32_f16(false, a, false, b, (short)0, c,
                                                false, false);
}

#define SWZ_XOR(x, m) \
  __int_as_float(__builtin_amdgcn_ds_swizzle(__float_as_int(x), ((m) << 10) | 0x1f))

// CDNA5 async DMA: copy 16B (or 2x16B) per lane from global to LDS, tracked by
// ASYNCcnt. INST_OFFSET is added to both the LDS and global addresses (ISA 8.4.4).
__device__ __forceinline__ void async_copy16(unsigned lds_off, const half_t* g) {
  asm volatile("global_load_async_to_lds_b128 %0, %1, off"
               :: "v"(lds_off), "v"((unsigned long long)(uintptr_t)g)
               : "memory");
}
__device__ __forceinline__ void async_copy32(unsigned lds_off, const half_t* g) {
  asm volatile("global_load_async_to_lds_b128 %0, %1, off\n\t"
               "global_load_async_to_lds_b128 %0, %1, off offset:16"
               :: "v"(lds_off), "v"((unsigned long long)(uintptr_t)g)
               : "memory");
}
__device__ __forceinline__ unsigned lds_off(const void* p) {
  return (unsigned)(uintptr_t)p;  // LDS aperture: addr[31:0] is the LDS offset
}
#define WAIT_ASYNC(n) asm volatile("s_wait_asynccnt " #n ::: "memory")

// ---------------- pack kernels (f32 -> f16, weight transpose) ----------------

__global__ void pack_x_kernel(const float* __restrict__ x,
                              half_t* __restrict__ xh, int n) {
  int i = blockIdx.x * blockDim.x + threadIdx.x;
  if (i < n) xh[i] = (half_t)x[i];
}

__global__ void pack_wqkv_kernel(const float* __restrict__ wq,
                                 const float* __restrict__ wk,
                                 const float* __restrict__ wv,
                                 half_t* __restrict__ wt) {
  int i = blockIdx.x * blockDim.x + threadIdx.x;  // over QKVN * Dd
  int row = i >> 11;                              // output feature (0..3071)
  int col = i & (Dd - 1);                         // k (0..2047)
  float v;
  if (row < Hh * HD)                 v = wq[(size_t)col * (Hh * HD) + row] * QSCALE;
  else if (row < Hh * HD + KVH * HD) v = wk[(size_t)col * (KVH * HD) + (row - Hh * HD)];
  else                               v = wv[(size_t)col * (KVH * HD) + (row - Hh * HD - KVH * HD)];
  wt[i] = (half_t)v;
}

__global__ void pack_wo_kernel(const float* __restrict__ wo,
                               half_t* __restrict__ wt) {
  int i = blockIdx.x * blockDim.x + threadIdx.x;  // over Dd * Dd
  int row = i >> 11;
  int col = i & (Dd - 1);
  wt[i] = (half_t)wo[(size_t)col * Dd + row];
}

// ---------------- tiled WMMA GEMM: C[M,N] = A[M,K] * Bt[N,K]^T ----------------
// Async-DMA double-buffered LDS staging. 8 waves, each computes a 32x64 strip.
// EPI==0: scatter to Q[b,h,n,d] / K[b,kvh,n,d] / Vt[b,kvh,d,n] (f16)
// EPI==1: dense f32 output row-major [M,Nout]

template <int EPI>
__global__ __launch_bounds__(256) void gemm_kernel(
    const half_t* __restrict__ A, const half_t* __restrict__ Bt,
    int M, int Nout, int Kdim,
    half_t* __restrict__ Qh, half_t* __restrict__ Kh, half_t* __restrict__ Vth,
    float* __restrict__ Cout) {
  constexpr int BM = 128, BN = 128, BK = 32;
  __shared__ __align__(16) half_t As[2][BM][BK + 8];  // stride 40 halves (80B)
  __shared__ __align__(16) half_t Bs[2][BN][BK + 8];

  const int tid   = threadIdx.x;
  const int wave  = tid >> 5;
  const int lane  = tid & 31;
  const int rl    = lane & 15;
  const int kb8   = (lane >> 4) * 8;
  const int waveM = wave >> 1;  // 0..3 -> 32-row strip
  const int waveN = wave & 1;   // 0..1 -> 64-col strip
  const int m0    = blockIdx.y * BM;
  const int n0    = blockIdx.x * BN;

  v8f acc[2][4];
#pragma unroll
  for (int i = 0; i < 2; ++i)
#pragma unroll
    for (int j = 0; j < 4; ++j)
#pragma unroll
      for (int e = 0; e < 8; ++e) acc[i][j][e] = 0.0f;

  const int lrow = tid >> 1;
  const int lcol = (tid & 1) * 16;
  const half_t* Ag = A + (size_t)(m0 + lrow) * Kdim + lcol;
  const half_t* Bg = Bt + (size_t)(n0 + lrow) * Kdim + lcol;

  // prologue: DMA tile k0=0 into ring slot 0 (4 async b128 per thread)
  async_copy32(lds_off(&As[0][lrow][lcol]), Ag);
  async_copy32(lds_off(&Bs[0][lrow][lcol]), Bg);

  int buf = 0;
  for (int k0 = 0; k0 < Kdim; k0 += BK) {
    if (k0 + BK < Kdim) {  // DMA next tile into the other ring slot
      async_copy32(lds_off(&As[buf ^ 1][lrow][lcol]), Ag + k0 + BK);
      async_copy32(lds_off(&Bs[buf ^ 1][lrow][lcol]), Bg + k0 + BK);
      WAIT_ASYNC(4);       // oldest 4 (current tile) landed; next 4 in flight
    } else {
      WAIT_ASYNC(0);
    }
    __syncthreads();       // all waves' current-tile DMA complete

    v16h af[2], bf[4];
#pragma unroll
    for (int i = 0; i < 2; ++i) af[i] = frag16(&As[buf][waveM * 32 + i * 16 + rl][kb8]);
#pragma unroll
    for (int j = 0; j < 4; ++j) bf[j] = frag16(&Bs[buf][waveN * 64 + j * 16 + rl][kb8]);
#pragma unroll
    for (int i = 0; i < 2; ++i)
#pragma unroll
      for (int j = 0; j < 4; ++j) acc[i][j] = wmma_f16(af[i], bf[j], acc[i][j]);

    __syncthreads();       // ring slot free for refill in the next iteration
    buf ^= 1;
  }

  // epilogue: C element (lane, vgpr r): row = base + r + 8*(lane>>4), col = base + lane%16
#pragma unroll
  for (int i = 0; i < 2; ++i)
#pragma unroll
    for (int j = 0; j < 4; ++j)
#pragma unroll
      for (int r = 0; r < 8; ++r) {
        const int gm = m0 + waveM * 32 + i * 16 + kb8 + r;
        const int gn = n0 + waveN * 64 + j * 16 + rl;
        const float v = acc[i][j][r];
        if constexpr (EPI == 1) {
          Cout[(size_t)gm * Nout + gn] = v;
        } else {
          const int bb = gm >> 11;         // batch
          const int nt = gm & (Nn - 1);    // token within batch
          if (gn < Hh * HD) {
            const int hh = gn >> 6, d = gn & 63;
            Qh[(((size_t)bb * Hh + hh) * Nn + nt) * HD + d] = (half_t)v;
          } else if (gn < Hh * HD + KVH * HD) {
            const int c = gn - Hh * HD;
            const int kvh = c >> 6, d = c & 63;
            Kh[(((size_t)bb * KVH + kvh) * Nn + nt) * HD + d] = (half_t)v;
          } else {
            const int c = gn - (Hh * HD + KVH * HD);
            const int kvh = c >> 6, d = c & 63;
            Vth[(((size_t)bb * KVH + kvh) * HD + d) * Nn + nt] = (half_t)v;  // V^T
          }
        }
      }
}

// ---------------- flash-attention: 1 block = (b, h, 128 query rows) ----------------

__global__ __launch_bounds__(256) void attn_kernel(
    const half_t* __restrict__ Q, const half_t* __restrict__ K,
    const half_t* __restrict__ Vt, half_t* __restrict__ O) {
  constexpr int KB = 32;  // keys per iteration
  __shared__ __align__(16) half_t Ks[2][KB][HD + 8];     // [32 keys][64 d], stride 72
  __shared__ __align__(16) half_t Vs[2][HD][KB + 8];     // [64 d][32 keys], stride 40
  __shared__ __align__(16) half_t Ps[8][16][KB + 8];     // per-wave P transpose scratch

  const int tid  = threadIdx.x;
  const int wave = tid >> 5;
  const int lane = tid & 31;
  const int rl   = lane & 15;
  const int kb8  = (lane >> 4) * 8;

  const int blk = blockIdx.x;
  const int qi  = blk & 15;          // N / 128 = 16 query blocks
  const int h   = (blk >> 4) & 31;
  const int b   = blk >> 9;
  const int kv  = h >> 2;            // GQA: repeat_interleave, rep = 4

  const int qrow = qi * 128 + wave * 16;

  // Q fragments held in registers for the whole key loop (A-frag layout).
  const half_t* Qp = Q + (((size_t)(b * Hh + h) * Nn) + qrow + rl) * HD;
  v16h qf0 = frag16(Qp + kb8);        // d = 0..31
  v16h qf1 = frag16(Qp + 32 + kb8);   // d = 32..63

  const half_t* Kg = K + ((size_t)(b * KVH + kv) * Nn) * HD;
  const half_t* Vg = Vt + ((size_t)(b * KVH + kv) * HD) * Nn;

  v8f o[4];
#pragma unroll
  for (int j = 0; j < 4; ++j)
#pragma unroll
    for (int e = 0; e < 8; ++e) o[j][e] = 0.0f;
  float m[8], l[8];
#pragma unroll
  for (int r = 0; r < 8; ++r) { m[r] = -3.0e38f; l[r] = 0.0f; }

  const int krow = tid >> 3, kcol = (tid & 7) * 8;  // K tile loader (32x64)
  const int vrow = tid >> 2, vcol = (tid & 3) * 8;  // V tile loader (64x32)

  // prologue: DMA tile kk=0 into ring slot 0 (2 async b128 per thread)
  async_copy16(lds_off(&Ks[0][krow][kcol]), Kg + (size_t)krow * HD + kcol);
  async_copy16(lds_off(&Vs[0][vrow][vcol]), Vg + (size_t)vrow * Nn + vcol);

  int buf = 0;
  for (int kk = 0; kk < Nn; kk += KB) {
    if (kk + KB < Nn) {  // DMA next K/V tile into the other ring slot
      async_copy16(lds_off(&Ks[buf ^ 1][krow][kcol]),
                   Kg + (size_t)(kk + KB + krow) * HD + kcol);
      async_copy16(lds_off(&Vs[buf ^ 1][vrow][vcol]),
                   Vg + (size_t)vrow * Nn + (kk + KB) + vcol);
      WAIT_ASYNC(2);
    } else {
      WAIT_ASYNC(0);
    }
    __syncthreads();

    // S = Q * K^T (scaled by 1/8*log2e via packed wq). Two 16x16 subtiles.
    v8f s0, s1;
#pragma unroll
    for (int e = 0; e < 8; ++e) { s0[e] = 0.0f; s1[e] = 0.0f; }
    s0 = wmma_f16(qf0, frag16(&Ks[buf][rl][kb8]), s0);
    s0 = wmma_f16(qf1, frag16(&Ks[buf][rl][32 + kb8]), s0);
    s1 = wmma_f16(qf0, frag16(&Ks[buf][16 + rl][kb8]), s1);
    s1 = wmma_f16(qf1, frag16(&Ks[buf][16 + rl][32 + kb8]), s1);

    // online softmax (base-2); row stats replicated across each 16-lane group
    float alpha[8];
#pragma unroll
    for (int r = 0; r < 8; ++r) {
      float t = fmaxf(s0[r], s1[r]);
      t = fmaxf(t, SWZ_XOR(t, 1));
      t = fmaxf(t, SWZ_XOR(t, 2));
      t = fmaxf(t, SWZ_XOR(t, 4));
      t = fmaxf(t, SWZ_XOR(t, 8));
      const float mn = fmaxf(m[r], t);
      const float a  = __builtin_amdgcn_exp2f(m[r] - mn);
      const float e0 = __builtin_amdgcn_exp2f(s0[r] - mn);
      const float e1 = __builtin_amdgcn_exp2f(s1[r] - mn);
      s0[r] = e0; s1[r] = e1;
      float rs = e0 + e1;
      rs += SWZ_XOR(rs, 1);
      rs += SWZ_XOR(rs, 2);
      rs += SWZ_XOR(rs, 4);
      rs += SWZ_XOR(rs, 8);
      l[r] = l[r] * a + rs;
      m[r] = mn;
      alpha[r] = a;
    }
#pragma unroll
    for (int j = 0; j < 4; ++j)
#pragma unroll
      for (int r = 0; r < 8; ++r) o[j][r] *= alpha[r];

    // C-layout P -> A-fragment layout via per-wave LDS round trip
#pragma unroll
    for (int r = 0; r < 8; ++r) {
      Ps[wave][kb8 + r][rl]      = (half_t)s0[r];
      Ps[wave][kb8 + r][16 + rl] = (half_t)s1[r];
    }
    asm volatile("s_wait_dscnt 0x0" ::: "memory");  // wave-local store->load fence
    v16h pf = frag16(&Ps[wave][rl][kb8]);
#pragma unroll
    for (int j = 0; j < 4; ++j)
      o[j] = wmma_f16(pf, frag16(&Vs[buf][j * 16 + rl][kb8]), o[j]);

    __syncthreads();   // ring slot free for refill
    buf ^= 1;
  }

  // normalize and store [b, n, h*64 + d] (f16) for the output projection
  half_t* Op = O + ((size_t)(b * Nn + qrow) * (Hh * HD)) + h * HD;
#pragma unroll
  for (int r = 0; r < 8; ++r) {
    const float inv = 1.0f / l[r];
    const int row = kb8 + r;
#pragma unroll
    for (int j = 0; j < 4; ++j)
      Op[(size_t)row * (Hh * HD) + j * 16 + rl] = (half_t)(o[j][r] * inv);
  }
}

}  // namespace

extern "C" void kernel_launch(void* const* d_in, const int* in_sizes, int n_in,
                              void* d_out, int out_size, void* d_ws, size_t ws_size,
                              hipStream_t stream) {
  (void)in_sizes; (void)n_in; (void)out_size; (void)ws_size;
  const float* x  = (const float*)d_in[0];
  const float* wq = (const float*)d_in[1];
  const float* wk = (const float*)d_in[2];
  const float* wv = (const float*)d_in[3];
  const float* wo = (const float*)d_in[4];
  float* out = (float*)d_out;
  char* ws = (char*)d_ws;

  // workspace layout (bytes), ~76 MB total
  half_t* xh    = (half_t*)(ws + 0);          // 16 MB  x in f16
  half_t* wqkvT = (half_t*)(ws + 16777216);   // 12 MB  [3072][2048] fused W^T
  half_t* woT   = (half_t*)(ws + 29360128);   //  8 MB  [2048][2048] wo^T
  half_t* Qh    = (half_t*)(ws + 37748736);   // 16 MB  [b][h][n][d]
  half_t* Kh    = (half_t*)(ws + 54525952);   //  4 MB  [b][kvh][n][d]
  half_t* Vth   = (half_t*)(ws + 58720256);   //  4 MB  [b][kvh][d][n]
  half_t* Ah    = (half_t*)(ws + 62914560);   // 16 MB  attn out [b][n][h*d]

  pack_x_kernel<<<(MTOK * Dd + 255) / 256, 256, 0, stream>>>(x, xh, MTOK * Dd);
  pack_wqkv_kernel<<<(QKVN * Dd) / 256, 256, 0, stream>>>(wq, wk, wv, wqkvT);
  pack_wo_kernel<<<(Dd * Dd) / 256, 256, 0, stream>>>(wo, woT);

  gemm_kernel<0><<<dim3(QKVN / 128, MTOK / 128), 256, 0, stream>>>(
      xh, wqkvT, MTOK, QKVN, Dd, Qh, Kh, Vth, nullptr);

  attn_kernel<<<Bb * Hh * (Nn / 128), 256, 0, stream>>>(Qh, Kh, Vth, Ah);

  gemm_kernel<1><<<dim3(Dd / 128, MTOK / 128), 256, 0, stream>>>(
      Ah, woT, MTOK, Dd, Dd, nullptr, nullptr, nullptr, out);
}
